// MultiHeadAttentionCombinedQKV_43903155700337
// MI455X (gfx1250) — matrix operand order repaired and checked by reading, not compiled, not run on hardware
//
#include <hip/hip_runtime.h>

typedef __attribute__((ext_vector_type(16))) __bf16       v16bf;
typedef __attribute__((ext_vector_type(8)))  __bf16       bf16x8;
typedef __attribute__((ext_vector_type(8)))  float        v8f;
typedef __attribute__((ext_vector_type(4)))  int          v4i;
typedef __attribute__((ext_vector_type(4)))  unsigned int u32x4;
typedef __attribute__((ext_vector_type(8)))  int          i32x8;
typedef __attribute__((ext_vector_type(4)))  int          i32x4;

#define BATCH  4
#define SEQ    2048
#define DMODEL 1024
#define NHEAD  16
#define HDIM   64

#if defined(__has_builtin)
#if __has_builtin(__builtin_amdgcn_global_load_async_to_lds_b128)
#define HAVE_ASYNC_BUILTIN 1
#endif
#if __has_builtin(__builtin_amdgcn_s_wait_asynccnt)
#define HAVE_WAIT_ASYNC_BUILTIN 1
#endif
#if __has_builtin(__builtin_amdgcn_tensor_load_to_lds)
#define HAVE_TDM 1
#endif
#if __has_builtin(__builtin_amdgcn_s_wait_tensorcnt)
#define HAVE_WAIT_TENSOR_BUILTIN 1
#endif
#endif

// Async builtin signature (probed round 2): (int4 AS1*, int4 AS3*, Ii, Ii)
typedef __attribute__((address_space(1))) v4i gbl_v4i_t;
typedef __attribute__((address_space(3))) v4i lds_v4i_t;

// Copy 16 bytes global -> LDS through the CDNA5 async path (ASYNCcnt).
__device__ __forceinline__ void async_copy16(void* lds_dst, const void* gsrc) {
#ifdef HAVE_ASYNC_BUILTIN
  __builtin_amdgcn_global_load_async_to_lds_b128((gbl_v4i_t*)gsrc,
                                                 (lds_v4i_t*)lds_dst, 0, 0);
#else
  unsigned l = (unsigned)(size_t)lds_dst;
  asm volatile("global_load_async_to_lds_b128 %0, %1, off" ::"v"(l), "v"(gsrc)
               : "memory");
#endif
}

__device__ __forceinline__ void wait_async() {
#ifdef HAVE_WAIT_ASYNC_BUILTIN
  __builtin_amdgcn_s_wait_asynccnt(0);
#else
  asm volatile("s_wait_asynccnt 0x0" ::: "memory");
#endif
}

__device__ __forceinline__ void wait_tensor() {
#ifdef HAVE_WAIT_TENSOR_BUILTIN
  __builtin_amdgcn_s_wait_tensorcnt(0);
#else
  asm volatile("s_wait_tensorcnt 0x0" ::: "memory");
#endif
}

#ifdef HAVE_TDM
// TDM 2D tile load: rows x cols(bf16) tile at gsrc (row stride = srcStrideElems,
// bf16 units) into LDS at lds_dst, padding each tile row in LDS.
// Descriptor packing per CDNA5 ISA 8.3/8.4 (group0: count/lds/global/type;
// group1: data_size=2B, pad, dims, tile, stride). Groups 2/3 zero (2D tensor).
// This toolchain's builtin is the 6-arg form: (g0, g1, g2, g3, g4, cpol).
__device__ __forceinline__ void tdm_load_tile_2d(void* lds_dst, const void* gsrc,
                                                 int rows, int cols,
                                                 int srcStrideElems,
                                                 unsigned padInterval,
                                                 unsigned padAmount) {
  unsigned lds_addr = (unsigned)(size_t)lds_dst;
  unsigned long long ga = (unsigned long long)(size_t)gsrc;
  u32x4 g0;
  g0[0] = 1u;                                         // count=1, user descriptor
  g0[1] = lds_addr;                                   // lds_addr
  g0[2] = (unsigned)(ga & 0xffffffffu);               // global_addr[31:0]
  g0[3] = (unsigned)((ga >> 32) & 0x01ffffffu) | (2u << 30);  // addr[56:32], type=2
  i32x8 g1;
  g1[0] = (int)((1u << 16) | (1u << 20) | (padInterval << 22) | (padAmount << 25));
  g1[1] = (int)((unsigned)cols << 16);                // tensor_dim0[15:0] @ bit48
  g1[2] = (int)((unsigned)rows << 16);                // tensor_dim1[15:0] @ bit80
  g1[3] = (int)((unsigned)cols << 16);                // tile_dim0 @ bit112
  g1[4] = rows;                                       // tile_dim1; tile_dim2=0
  g1[5] = srcStrideElems;                             // tensor_dim0_stride[31:0]
  g1[6] = 0;
  g1[7] = 0;
  i32x4 z4;
  z4[0] = 0; z4[1] = 0; z4[2] = 0; z4[3] = 0;
  i32x8 z8;
  z8[0] = 0; z8[1] = 0; z8[2] = 0; z8[3] = 0;
  z8[4] = 0; z8[5] = 0; z8[6] = 0; z8[7] = 0;
  __builtin_amdgcn_tensor_load_to_lds(g0, g1, z4, z4, z8, 0);
}
#endif

__device__ __forceinline__ v8f wmma_bf16f32(v16bf a, v16bf b, v8f c) {
  return __builtin_amdgcn_wmma_f32_16x16x32_bf16(false, a, false, b, (short)0, c,
                                                 false, false);
}

// Build a 16-element bf16 fragment from two 16-byte-aligned 8-element chunks.
// A-fragments: off2=16 (K chunks at +0 / +16). B-fragments: off2=16 too (same K order as A).
__device__ __forceinline__ v16bf frag_from(const __bf16* p, int off2) {
  bf16x8 lo = *(const bf16x8*)p;
  bf16x8 hi = *(const bf16x8*)(p + off2);
  v16bf r;
#pragma unroll
  for (int i = 0; i < 8; ++i) { r[i] = lo[i]; r[i + 8] = hi[i]; }
  return r;
}

__global__ void cvt_f32_bf16(const float* __restrict__ in, __bf16* __restrict__ out, int n) {
  for (int i = blockIdx.x * blockDim.x + threadIdx.x; i < n; i += gridDim.x * blockDim.x) {
    float v = __builtin_nontemporal_load(&in[i]);  // streaming pass: keep L2 clean
    __builtin_nontemporal_store((__bf16)v, &out[i]);
  }
}

// C[M,N] = A[M,K](bf16,rm) * B[K,N](bf16,rm). 64x128 tile per WG (4 waves, each a
// 64x32 strip -> 8 WMMAs per K-step). A staged via async-to-LDS; B transposed in LDS.
__global__ __launch_bounds__(128) void gemm_bf16_wmma(
    const __bf16* __restrict__ A, const __bf16* __restrict__ B,
    int K, int lda, int ldb, int ldc,
    __bf16* __restrict__ Cbf, float* __restrict__ Cf, const float* __restrict__ bias) {
  __shared__ __bf16 sA[64][40];    // 64 x 32 K (+8 pad), row stride 80B
  __shared__ __bf16 sBT[128][40];  // transposed [n][k] (+8 pad)
  const int tileM = blockIdx.y * 64, tileN = blockIdx.x * 128;
  const int wave = threadIdx.x >> 5, lane = threadIdx.x & 31;
  const int lrow = lane & 15, khalf = lane >> 4;
  const int nBase = wave * 32;

  v8f acc[4][2];
#pragma unroll
  for (int t = 0; t < 4; ++t)
#pragma unroll
    for (int n = 0; n < 2; ++n)
#pragma unroll
      for (int r = 0; r < 8; ++r) acc[t][n][r] = 0.0f;

  for (int k0 = 0; k0 < K; k0 += 32) {
    __syncthreads();
    for (int i = threadIdx.x; i < 512; i += 128) {
      if (i < 256) {  // A tile 64x32, async global->LDS
        int r = (i * 8) / 32, c = (i * 8) % 32;
        const __bf16* ap = A + (size_t)(tileM + r) * lda + k0 + c;
        async_copy16(&sA[r][c], ap);
        if (k0 + 32 < K) __builtin_prefetch(ap + 32, 0, 0);
      }
      {  // B tile 32x128, load + transpose-store
        int r = (i * 8) / 128, c = (i * 8) % 128;
        const __bf16* bp = B + (size_t)(k0 + r) * ldb + tileN + c;
        bf16x8 v = *(const bf16x8*)bp;
        if (k0 + 32 < K) __builtin_prefetch(bp + (size_t)32 * ldb, 0, 0);
#pragma unroll
        for (int j = 0; j < 8; ++j) sBT[c + j][r] = v[j];
      }
    }
    wait_async();
    __syncthreads();

    v16bf b0 = frag_from(&sBT[nBase + lrow][8 * khalf], 16);
    v16bf b1 = frag_from(&sBT[nBase + 16 + lrow][8 * khalf], 16);
#pragma unroll
    for (int mi = 0; mi < 4; ++mi) {
      v16bf af = frag_from(&sA[mi * 16 + lrow][8 * khalf], 16);
      acc[mi][0] = wmma_bf16f32(af, b0, acc[mi][0]);
      acc[mi][1] = wmma_bf16f32(af, b1, acc[mi][1]);
    }
  }

#pragma unroll
  for (int mi = 0; mi < 4; ++mi)
#pragma unroll
    for (int ni = 0; ni < 2; ++ni)
#pragma unroll
      for (int r = 0; r < 8; ++r) {
        int row = tileM + mi * 16 + r + 8 * khalf;
        int col = tileN + nBase + ni * 16 + lrow;
        if (Cf) Cf[(size_t)row * ldc + col] = acc[mi][ni][r] + (bias ? bias[col] : 0.0f);
        else    Cbf[(size_t)row * ldc + col] = (__bf16)acc[mi][ni][r];
      }
}

// Flash attention, causal. Grid: B*H*(S/64) WGs of 128 threads.
// K block staged by the Tensor Data Mover (TENSORcnt); V transposed manually.
__global__ __launch_bounds__(128) void attn_flash_wmma(
    const __bf16* __restrict__ qkv, __bf16* __restrict__ ctx) {
  const int qblocks = SEQ / 64;
  const int qb = blockIdx.x % qblocks;
  const int h  = (blockIdx.x / qblocks) % NHEAD;
  const int b  = blockIdx.x / (qblocks * NHEAD);

  const int wave = threadIdx.x >> 5, lane = threadIdx.x & 31;
  const int lrow = lane & 15, khalf = lane >> 4;
  const int qRow0 = qb * 64 + wave * 16;
  const int rs = 3 * DMODEL;

  const __bf16* qbase = qkv + (size_t)b * SEQ * rs + h * HDIM;
  const __bf16* kbase = qbase + DMODEL;
  const __bf16* vbase = qbase + 2 * DMODEL;

  __shared__ __bf16 ldsK[64][80];      // [keypos][hd], 128B data + 32B pad per row
  __shared__ __bf16 ldsVT[64][80];     // [hd][keypos]
  __shared__ __bf16 ldsP[4][16][80];   // per-wave P tile [qrow][keypos]

  v16bf qf[2];  // q strip, pre-scaled by 1/sqrt(HD)
#pragma unroll
  for (int s = 0; s < 2; ++s) {
    const __bf16* qp = qbase + (size_t)(qRow0 + lrow) * rs + s * 32 + 8 * khalf;
    bf16x8 lo = *(const bf16x8*)qp;
    bf16x8 hi = *(const bf16x8*)(qp + 16);
#pragma unroll
    for (int e = 0; e < 8; ++e) {
      qf[s][e]     = (__bf16)((float)lo[e] * 0.125f);
      qf[s][e + 8] = (__bf16)((float)hi[e] * 0.125f);
    }
  }

  v8f acc[4];
  float m_i[8], l_i[8];
#pragma unroll
  for (int t = 0; t < 4; ++t)
#pragma unroll
    for (int r = 0; r < 8; ++r) acc[t][r] = 0.0f;
#pragma unroll
  for (int r = 0; r < 8; ++r) { m_i[r] = -1e30f; l_i[r] = 0.0f; }

  for (int kb = 0; kb <= qb; ++kb) {
    const int key0 = kb * 64;
    __syncthreads();
#ifdef HAVE_TDM
    if (wave == 0) {  // one wave drives the TDM; tile row 128B -> interval code 4,
                      // pad 8 dwords -> amount code 7 (matches ldsK row pad)
      tdm_load_tile_2d(&ldsK[0][0], kbase + (size_t)key0 * rs, 64, HDIM, rs, 4u, 7u);
    }
#endif
    for (int i = threadIdx.x; i < 512; i += 128) {
      int r = i >> 3, c = (i & 7) * 8;
#ifndef HAVE_TDM
      async_copy16(&ldsK[r][c], kbase + (size_t)(key0 + r) * rs + c);
#endif
      const __bf16* vp = vbase + (size_t)(key0 + r) * rs + c;
      bf16x8 vv = *(const bf16x8*)vp;
      if (kb < qb) __builtin_prefetch(vp + (size_t)64 * rs, 0, 0);
#pragma unroll
      for (int j = 0; j < 8; ++j) ldsVT[c + j][r] = vv[j];
    }
#ifdef HAVE_TDM
    if (wave == 0) wait_tensor();
#else
    wait_async();
#endif
    __syncthreads();

    v8f sc[4];
#pragma unroll
    for (int t = 0; t < 4; ++t) {
      v8f z;
#pragma unroll
      for (int r = 0; r < 8; ++r) z[r] = 0.0f;
#pragma unroll
      for (int s = 0; s < 2; ++s) {
        v16bf bfr = frag_from(&ldsK[t * 16 + lrow][s * 32 + 8 * khalf], 16);
        z = wmma_bf16f32(qf[s], bfr, z);
      }
      sc[t] = z;
    }

    if (key0 + 63 > qRow0) {  // causal mask on diagonal blocks
#pragma unroll
      for (int t = 0; t < 4; ++t)
#pragma unroll
        for (int r = 0; r < 8; ++r)
          if (key0 + t * 16 + lrow > qRow0 + r + 8 * khalf) sc[t][r] = -1e30f;
    }

    float scl[8], rsum[8];
#pragma unroll
    for (int r = 0; r < 8; ++r) {
      float v = fmaxf(fmaxf(sc[0][r], sc[1][r]), fmaxf(sc[2][r], sc[3][r]));
      v = fmaxf(v, __shfl_xor(v, 1, 32));
      v = fmaxf(v, __shfl_xor(v, 2, 32));
      v = fmaxf(v, __shfl_xor(v, 4, 32));
      v = fmaxf(v, __shfl_xor(v, 8, 32));
      float mnew = fmaxf(m_i[r], v);
      scl[r] = __expf(m_i[r] - mnew);
      m_i[r] = mnew;
      rsum[r] = 0.0f;
    }
#pragma unroll
    for (int t = 0; t < 4; ++t)
#pragma unroll
      for (int r = 0; r < 8; ++r) {
        float p = __expf(sc[t][r] - m_i[r]);
        sc[t][r] = p;
        rsum[r] += p;
      }
#pragma unroll
    for (int r = 0; r < 8; ++r) {
      float v = rsum[r];
      v += __shfl_xor(v, 1, 32);
      v += __shfl_xor(v, 2, 32);
      v += __shfl_xor(v, 4, 32);
      v += __shfl_xor(v, 8, 32);
      l_i[r] = l_i[r] * scl[r] + v;
#pragma unroll
      for (int t = 0; t < 4; ++t) acc[t][r] *= scl[r];
    }

#pragma unroll
    for (int t = 0; t < 4; ++t)
#pragma unroll
      for (int r = 0; r < 8; ++r)
        ldsP[wave][r + 8 * khalf][t * 16 + lrow] = (__bf16)sc[t][r];

#pragma unroll
    for (int s = 0; s < 2; ++s) {
      v16bf af = frag_from(&ldsP[wave][lrow][s * 32 + 8 * khalf], 16);
#pragma unroll
      for (int t = 0; t < 4; ++t) {
        v16bf bfr = frag_from(&ldsVT[t * 16 + lrow][s * 32 + 8 * khalf], 16);
        acc[t] = wmma_bf16f32(af, bfr, acc[t]);
      }
    }
  }

  __bf16* op = ctx + (size_t)b * SEQ * DMODEL + h * HDIM;
#pragma unroll
  for (int r = 0; r < 8; ++r) {
    float inv = 1.0f / l_i[r];
    int row = qRow0 + r + 8 * khalf;
#pragma unroll
    for (int t = 0; t < 4; ++t)
      op[(size_t)row * DMODEL + t * 16 + lrow] = (__bf16)(acc[t][r] * inv);
  }
}

extern "C" void kernel_launch(void* const* d_in, const int* in_sizes, int n_in,
                              void* d_out, int out_size, void* d_ws, size_t ws_size,
                              hipStream_t stream) {
  (void)in_sizes; (void)n_in; (void)out_size; (void)ws_size;
  const float* x    = (const float*)d_in[0];  // [B,S,D]
  const float* Wqkv = (const float*)d_in[1];  // [D,3D]
  const float* Wout = (const float*)d_in[2];  // [D,D]
  const float* bout = (const float*)d_in[3];  // [D]
  float* out = (float*)d_out;                 // [B,S,D]

  const size_t MR = (size_t)BATCH * SEQ;  // 8192 rows
  char* ws = (char*)d_ws;
  size_t off = 0;
  __bf16* xb   = (__bf16*)(ws + off); off += MR * DMODEL * 2;
  __bf16* wqb  = (__bf16*)(ws + off); off += (size_t)DMODEL * 3 * DMODEL * 2;
  __bf16* wob  = (__bf16*)(ws + off); off += (size_t)DMODEL * DMODEL * 2;
  __bf16* qkvb = (__bf16*)(ws + off); off += MR * 3 * DMODEL * 2;
  __bf16* ctxb = (__bf16*)(ws + off);

  cvt_f32_bf16<<<2048, 256, 0, stream>>>(x,    xb,  (int)(MR * DMODEL));
  cvt_f32_bf16<<<2048, 256, 0, stream>>>(Wqkv, wqb, DMODEL * 3 * DMODEL);
  cvt_f32_bf16<<<1024, 256, 0, stream>>>(Wout, wob, DMODEL * DMODEL);

  dim3 g1(3 * DMODEL / 128, (unsigned)(MR / 64));
  gemm_bf16_wmma<<<g1, 128, 0, stream>>>(xb, wqb, DMODEL, DMODEL, 3 * DMODEL,
                                         3 * DMODEL, qkvb, nullptr, nullptr);

  attn_flash_wmma<<<BATCH * NHEAD * (SEQ / 64), 128, 0, stream>>>(qkvb, ctxb);

  dim3 g2(DMODEL / 128, (unsigned)(MR / 64));
  gemm_bf16_wmma<<<g2, 128, 0, stream>>>(ctxb, wob, DMODEL, DMODEL, DMODEL,
                                         DMODEL, nullptr, out, bout);
}
